// NextFrameClassifier_12489764897539
// MI455X (gfx1250) — compile-verified, matrix operand order checked by reading, not run on hardware
//
#include <hip/hip_runtime.h>
#include <hip/hip_bf16.h>

typedef __attribute__((ext_vector_type(16))) _Float16 v16h;
typedef __attribute__((ext_vector_type(8)))  float    v8f;

#define SCALE_F 100000.0f
#define THRES_F 0.04f

__device__ inline v8f z8() {
  v8f z = {0.f,0.f,0.f,0.f,0.f,0.f,0.f,0.f};
  return z;
}

// Load a 16x32 f16 fragment from row-major [*, ld] memory, base at (row0,k0).
// Per CDNA5 16-bit A layout: lanes 0-15 -> row=lane, K 0..7 & 16..23;
// lanes 16-31 -> row=lane-16, K 8..15 & 24..31.
__device__ inline v16h load_frag(const _Float16* base, int ld) {
  int lane = threadIdx.x & 31;
  int r = lane & 15;
  int kh = (lane >> 4) << 3;             // 0 or 8
  const _Float16* p = base + (size_t)r * ld + kh;
  union { v16h v; uint4 q[2]; } u;
  u.q[0] = *(const uint4*)(p);
  u.q[1] = *(const uint4*)(p + 16);
  return u.v;
}

__device__ inline v8f wmma32(v16h a, v16h b, v8f c) {
  return __builtin_amdgcn_wmma_f32_16x16x32_f16(false, a, false, b, (short)0, c,
                                                false, false);
}

__device__ inline float sigf(float x) { return 1.f / (1.f + expf(-x)); }

// ---------------------------------------------------------------- conversions
__global__ void k_cvt(const float* __restrict__ src, _Float16* __restrict__ dst,
                      int R, int Cs, int Cd) {
  size_t idx = (size_t)blockIdx.x * blockDim.x + threadIdx.x;
  size_t tot = (size_t)R * Cd;
  if (idx >= tot) return;
  int r = (int)(idx / Cd), c = (int)(idx % Cd);
  dst[idx] = (c < Cs) ? (_Float16)src[(size_t)r * Cs + c] : (_Float16)0.f;
}

__global__ void k_convw(const float* __restrict__ src, _Float16* __restrict__ dst) {
  int idx = blockIdx.x * 256 + threadIdx.x;          // [5][256][256]
  if (idx >= 5 * 256 * 256) return;
  int c = idx & 255, o = (idx >> 8) & 255, k = idx >> 16;
  dst[idx] = (_Float16)src[((size_t)o * 256 + c) * 5 + k];
}

__global__ void k_zero16(_Float16* p, size_t n) {
  size_t i = (size_t)blockIdx.x * blockDim.x + threadIdx.x;
  if (i < n) p[i] = (_Float16)0.f;
}

// ------------------------------------------- generic WMMA GEMM, NT-wide N-blocking
// C[M,N] = A[M,K] @ B[N,K]^T + bias ; blockDim=128 (4 waves), wave = 16 x (16*NT)
template <int NT>
__global__ void k_gemm(const _Float16* __restrict__ A, int lda,
                       const _Float16* __restrict__ Bm, int ldb,
                       const float* __restrict__ bias,
                       float* __restrict__ C, int ldc, int M, int N, int K) {
  int wave = threadIdx.x >> 5;
  int m0 = (blockIdx.y * 4 + wave) * 16;
  int n0 = blockIdx.x * (16 * NT);
  if (m0 >= M || n0 >= N) return;
  v8f acc[NT];
#pragma unroll
  for (int j = 0; j < NT; j++) acc[j] = z8();
  const _Float16* ap = A + (size_t)m0 * lda;
  const _Float16* bp = Bm + (size_t)n0 * ldb;
  for (int k0 = 0; k0 < K; k0 += 32) {
    __builtin_prefetch(ap + k0 + 32, 0, 1);
    v16h a = load_frag(ap + k0, lda);
#pragma unroll
    for (int j = 0; j < NT; j++) {
      v16h b = load_frag(bp + (size_t)(j * 16) * ldb + k0, ldb);
      acc[j] = wmma32(a, b, acc[j]);
    }
  }
  int lane = threadIdx.x & 31;
  int n = lane & 15, mh = (lane >> 4) << 3;
#pragma unroll
  for (int j = 0; j < NT; j++) {
    int nc = n0 + j * 16 + n;
    float bv = bias ? bias[nc] : 0.f;
    for (int r = 0; r < 8; r++) {
      int m = m0 + mh + r;
      if (m < M) C[(size_t)m * ldc + nc] = acc[j][r] + bv;
    }
  }
}

// ------------------------------------------------ persistent encoder GRU (WMMA)
// One workgroup, 16 waves. h[16,256] f16 in LDS. gix = precomputed x@Wi.T+bi.
// Wh is streamed from L2 every step: an opaque integer offset defeats LICM (so
// the 24 fragments are not hoisted+spilled) while keeping the kernel-arg pointer
// visible so the loads stay on the global_load path (separate LOADcnt tracking).
__global__ void __launch_bounds__(512, 1)
k_encgru(const float* __restrict__ gix,
         const _Float16* __restrict__ Wh16,
         const float* __restrict__ bh,
         float* __restrict__ gru32,
         _Float16* __restrict__ gru16,
         _Float16* __restrict__ gt16) {
  __shared__ _Float16 hsh[16 * 256];
  int tid = threadIdx.x;
  for (int i = tid; i < 16 * 256; i += 512) hsh[i] = (_Float16)0.f;
  __syncthreads();
  int wave = tid >> 5, lane = tid & 31;
  int cw = wave * 16;                     // column slice within each gate
  int n = lane & 15, mh = (lane >> 4) << 3;
  int col = cw + n;
  float bhr = bh[col], bhz = bh[256 + col], bhn = bh[512 + col];
  for (int t = 0; t < 1024; t++) {
    unsigned toff = 0;
    asm volatile("" : "+s"(toff));         // defeat LICM: stream Wh from L2
    const _Float16* whp = Wh16 + toff;
    v8f a0 = z8(), a1 = z8(), a2 = z8();
    for (int k0 = 0; k0 < 256; k0 += 32) {
      v16h af = load_frag(hsh + k0, 256);
      v16h b0 = load_frag(whp + (size_t)(cw) * 256 + k0, 256);
      v16h b1 = load_frag(whp + (size_t)(256 + cw) * 256 + k0, 256);
      v16h b2 = load_frag(whp + (size_t)(512 + cw) * 256 + k0, 256);
      a0 = wmma32(af, b0, a0);
      a1 = wmma32(af, b1, a1);
      a2 = wmma32(af, b2, a2);
    }
    __syncthreads();                       // all waves done reading hsh
    for (int r = 0; r < 8; r++) {
      int m = mh + r;                      // batch row
      size_t grow = (size_t)m * 1024 + t;
      float gir = gix[grow * 768 + col];
      float giz = gix[grow * 768 + 256 + col];
      float gin = gix[grow * 768 + 512 + col];
      float rr = sigf(gir + a0[r] + bhr);
      float zz = sigf(giz + a1[r] + bhz);
      float nn = tanhf(gin + rr * (a2[r] + bhn));
      float hp = (float)hsh[m * 256 + col];
      float hv = (1.f - zz) * nn + zz * hp;
      gru32[grow * 256 + col] = hv;
      gru16[grow * 256 + col] = (_Float16)hv;
      gt16[((size_t)m * 256 + col) * 1024 + t] = (_Float16)hv;
      hsh[m * 256 + col] = (_Float16)hv;   // unique (m,col) per lane post-barrier
    }
    __syncthreads();
  }
}

// -------------------------------------------------- adjacent cosine similarity
// mode 0: write sim[b,t] (D=256 gru_out). mode 1: write preds pos/neg (D=64 latent).
__global__ void k_cos(const float* __restrict__ x, int D, int T, int mode,
                      float* __restrict__ sim_out, float* __restrict__ preds_out) {
  int gw = (blockIdx.x * blockDim.x + threadIdx.x) >> 5;
  int lane = threadIdx.x & 31;
  int nt = T - 1;
  if (gw >= 16 * nt) return;
  int b = gw / nt, t = gw % nt;
  const float* a = x + ((size_t)b * T + t) * D;
  const float* bb = a + D;
  float aa = 0.f, ab = 0.f, b2 = 0.f;
  for (int i = lane; i < D; i += 32) {
    float av = a[i], bv = bb[i];
    aa += av * av; ab += av * bv; b2 += bv * bv;
  }
  for (int o = 16; o >= 1; o >>= 1) {
    aa += __shfl_xor(aa, o, 32);
    ab += __shfl_xor(ab, o, 32);
    b2 += __shfl_xor(b2, o, 32);
  }
  if (lane == 0) {
    float na = fmaxf(sqrtf(aa), 1e-6f), nb = fmaxf(sqrtf(b2), 1e-6f);
    float cs = ab / (na * nb);
    if (mode == 0) sim_out[(size_t)b * nt + t] = cs;
    else {
      preds_out[((size_t)b * nt + t) * 2 + 0] = cs;            // pos
      preds_out[((size_t)b * nt + t) * 2 + 1] = aa / (na * na); // neg = cos(a,a)
    }
  }
}

__global__ void k_minmax(const float* __restrict__ sim, const int* __restrict__ len,
                         float* __restrict__ smm) {
  int b = blockIdx.x;
  __shared__ float smn[256], smx[256];
  float mn = __builtin_inff(), mx = -__builtin_inff();
  int L = len[b];
  for (int t = threadIdx.x; t < 1023; t += 256) {
    if (t + 1 < L) { float s = sim[b * 1023 + t]; mn = fminf(mn, s); mx = fmaxf(mx, s); }
  }
  smn[threadIdx.x] = mn; smx[threadIdx.x] = mx;
  __syncthreads();
  for (int o = 128; o > 0; o >>= 1) {
    if (threadIdx.x < o) {
      smn[threadIdx.x] = fminf(smn[threadIdx.x], smn[threadIdx.x + o]);
      smx[threadIdx.x] = fmaxf(smx[threadIdx.x], smx[threadIdx.x + o]);
    }
    __syncthreads();
  }
  if (threadIdx.x == 0) { smm[b * 2] = smn[0]; smm[b * 2 + 1] = smx[0]; }
}

__global__ void k_d(const float* __restrict__ sim, const int* __restrict__ len,
                    const float* __restrict__ smm, float* __restrict__ dbuf,
                    float* __restrict__ od, float* __restrict__ omask) {
  int idx = blockIdx.x * 256 + threadIdx.x;
  if (idx >= 16 * 1024) return;
  int b = idx >> 10, t = idx & 1023;
  int L = len[b];
  float dv = 0.f;
  if (t < 1023 && (t + 1) < L) {
    float mn = smm[2 * b], mx = smm[2 * b + 1];
    dv = 1.f - (sim[b * 1023 + t] - mn) / (mx - mn + 1e-6f);
  }
  dbuf[idx] = dv;
  od[idx] = dv;
  omask[idx] = (t < L) ? 1.f : 0.f;
}

__global__ void k_pb(const float* __restrict__ dbuf, const int* __restrict__ len,
                     float* __restrict__ pbbuf, float* __restrict__ opb) {
  int idx = blockIdx.x * 256 + threadIdx.x;
  if (idx >= 16 * 1024) return;
  int b = idx >> 10, t = idx & 1023;
  int L = len[b];
  bool dmt = (t < 1023) && (t + 1 < L);
  float dv = dbuf[idx];
  float left  = dmt ? ((t >= 1) ? dbuf[idx - 1] : 0.f) : 0.f;
  float right = (t < 1023) ? dbuf[idx + 1] : 0.f;
  float left2  = dmt ? ((t >= 2) ? dbuf[idx - 2] : 0.f) : 0.f;
  float right2 = (t < 1022) ? dbuf[idx + 2] : 0.f;
  float p1 = fminf(fmaxf(dv - left, 0.f), fmaxf(dv - right, 0.f));
  float p2 = fminf(fmaxf(dv - left2, 0.f), fmaxf(dv - right2, 0.f));
  float p = fminf(fmaxf(fmaxf(p1, p2) - THRES_F, 0.f), p1);
  float ps = tanhf(10.f * p);
  float ph = tanhf(fminf(1e9f * p, 30.f));
  float pbv = ps + (ph - ps);              // straight-through fwd value
  pbbuf[idx] = pbv;
  opb[idx] = pbv;
}

__global__ void k_cumsum(const float* __restrict__ pb, const int* __restrict__ len,
                         float* __restrict__ acc) {
  int b = threadIdx.x;
  if (b >= 16) return;
  float s = 0.f;
  int L = len[b];
  for (int t = 0; t < 1024; t++) {
    s += pb[b * 1024 + t];
    acc[b * 1024 + t] = (t < L) ? s : __builtin_inff();
  }
}

__global__ void k_dur(const float* __restrict__ acc, int* __restrict__ duri,
                      float* __restrict__ wdiv, float* __restrict__ odur) {
  int gm = blockIdx.x;                     // [0, 16*1025)
  int b = gm / 1025, m = gm % 1025;
  __shared__ float red[256];
  float s = 0.f;
  float rm = (float)m;
  for (int t = threadIdx.x; t < 1024; t += 256) {
    float v = rm - acc[b * 1024 + t];
    s += 1.f - tanhf(fminf(SCALE_F * fabsf(v), 12.f));
  }
  red[threadIdx.x] = s;
  __syncthreads();
  for (int o = 128; o > 0; o >>= 1) {
    if (threadIdx.x < o) red[threadIdx.x] += red[threadIdx.x + o];
    __syncthreads();
  }
  if (threadIdx.x == 0) {
    int di = (int)red[0];
    duri[gm] = di;
    wdiv[gm] = 1.f / ((float)di + 1e-6f);
    odur[gm] = (float)di;
  }
}

// ----------------- segment pooling: seg[b] = W[b](on-the-fly f16) @ G^T[b]
// B operand (G^T k-chunks, 16KB) staged global->LDS with double-buffered
// async copies (ASYNCcnt), consumed as ds_load fragments by all 8 waves.
__global__ void __launch_bounds__(256, 1)
k_seg(const float* __restrict__ acc, const float* __restrict__ wdiv,
      const _Float16* __restrict__ gt, _Float16* __restrict__ segbuf) {
  __shared__ float sa[1024];
  __shared__ _Float16 gsh[2][256 * 32];    // [buf][z][k] k-chunk of G^T, 16KB each
  int b = blockIdx.x;
  int tid = threadIdx.x;
  const _Float16* gtb = gt + (size_t)b * 256 * 1024;
  for (int t = tid; t < 1024; t += 256) sa[t] = acc[b * 1024 + t];

  // async issue of one 256x32 chunk: 4 x b128 per thread
  auto issue = [&](int buf, int k0) {
    int p = tid & 3;
    int zb = tid >> 2;
#pragma unroll
    for (int i = 0; i < 4; ++i) {
      int z = zb + 64 * i;
      unsigned loff = (unsigned)(size_t)(const void*)&gsh[buf][z * 32 + p * 8];
      const _Float16* gp = gtb + (size_t)z * 1024 + k0 + p * 8;
      asm volatile("global_load_async_to_lds_b128 %0, %1, off"
                   :: "v"(loff), "v"(gp) : "memory");
    }
  };

  issue(0, 0);
  __syncthreads();                          // sa ready

  int wave = tid >> 5, lane = tid & 31;
  int mt = blockIdx.y * 8 + wave;           // M = 1025 -> 65 tiles (padded 1040)
  int m0 = mt * 16;
  int mrow = m0 + (lane & 15);
  float rm = (float)mrow;
  float rd = (mrow < 1025) ? wdiv[b * 1025 + mrow] : 0.f;

  v8f accv[16];
#pragma unroll
  for (int j = 0; j < 16; j++) accv[j] = z8();

  for (int c = 0; c < 32; ++c) {
    int cur = c & 1;
    if (c + 1 < 32) {
      issue(cur ^ 1, (c + 1) * 32);
      asm volatile("s_wait_asynccnt 4" ::: "memory");   // chunk c landed
    } else {
      asm volatile("s_wait_asynccnt 0" ::: "memory");
    }
    __syncthreads();                        // visible to all waves

    int tb = c * 32 + ((lane >> 4) << 3);
    union { v16h v; _Float16 h[16]; } af;
#pragma unroll
    for (int i = 0; i < 8; i++) {
      float v1 = rm - sa[tb + i];
      af.h[i] = (_Float16)((1.f - tanhf(fminf(SCALE_F * fabsf(v1), 12.f))) * rd);
      float v2 = rm - sa[tb + 16 + i];
      af.h[8 + i] = (_Float16)((1.f - tanhf(fminf(SCALE_F * fabsf(v2), 12.f))) * rd);
    }
    const _Float16* gp = &gsh[cur][0];
#pragma unroll
    for (int j = 0; j < 16; j++) {
      v16h bf = load_frag(gp + (size_t)(j * 16) * 32, 32);
      accv[j] = wmma32(af.v, bf, accv[j]);
    }
    __syncthreads();                        // reads done before chunk c+2 lands
  }

  if (mt < 65) {
    _Float16* sb = segbuf + (size_t)b * 1048 * 256;
    int n = lane & 15, mh = (lane >> 4) << 3;
#pragma unroll
    for (int j = 0; j < 16; j++)
      for (int r = 0; r < 8; r++) {
        int m = m0 + mh + r;
        float v = (m < 1025) ? accv[j][r] : 0.f;
        sb[(size_t)(2 + m) * 256 + j * 16 + n] = (_Float16)v;
      }
  }
}

// --------------------------- conv1d K=5 pad=2 as 5 shift-accumulated WMMA GEMMs
__global__ void __launch_bounds__(256, 1)
k_conv(const _Float16* __restrict__ segbuf,
       const _Float16* __restrict__ wconv,
       const float* __restrict__ cb, float* __restrict__ out) {
  int b = blockIdx.x;
  int wave = threadIdx.x >> 5, lane = threadIdx.x & 31;
  int mt = blockIdx.y * 8 + wave;
  if (mt >= 65) return;
  int m0 = mt * 16;
  const _Float16* sb = segbuf + (size_t)b * 1048 * 256;
  v8f accv[16];
#pragma unroll
  for (int j = 0; j < 16; j++) accv[j] = z8();
  for (int tap = 0; tap < 5; tap++) {
    const _Float16* ab = sb + (size_t)(m0 + tap) * 256;   // (2 + m0 + tap - 2)
    const _Float16* wb = wconv + (size_t)tap * 256 * 256;
    for (int k0 = 0; k0 < 256; k0 += 32) {
      v16h af = load_frag(ab + k0, 256);
#pragma unroll
      for (int j = 0; j < 16; j++) {
        v16h bf = load_frag(wb + (size_t)(j * 16) * 256 + k0, 256);
        accv[j] = wmma32(af, bf, accv[j]);
      }
    }
  }
  int n = lane & 15, mh = (lane >> 4) << 3;
#pragma unroll
  for (int j = 0; j < 16; j++) {
    int nn = j * 16 + n;
    float bias = cb[nn];
    for (int r = 0; r < 8; r++) {
      int m = m0 + mh + r;
      if (m < 1025) out[((size_t)b * 1025 + m) * 256 + nn] = accv[j][r] + bias;
    }
  }
}

__global__ void k_prefix(const int* __restrict__ duri, int* __restrict__ starts) {
  if (blockIdx.x == 0 && threadIdx.x == 0) {
    long total = 0;
    for (int i = 0; i < 16400; i++) {
      starts[i] = (int)(total < 16384 ? total : 16384);
      int d = duri[i];
      total += (d > 0) ? d : 0;
    }
    starts[16400] = 16384;
  }
}

__global__ void k_up(const int* __restrict__ starts, const float* __restrict__ segrep,
                     float* __restrict__ frame, _Float16* __restrict__ frame16) {
  int j = blockIdx.x;
  __shared__ int si;
  if (threadIdx.x == 0) {
    int lo = 0, hi = 16399;
    while (lo < hi) {                      // largest i with starts[i] <= j
      int mid = (lo + hi + 1) >> 1;
      if (starts[mid] <= j) lo = mid; else hi = mid - 1;
    }
    si = lo;
  }
  __syncthreads();
  float v = segrep[(size_t)si * 256 + threadIdx.x];
  frame[(size_t)j * 256 + threadIdx.x] = v;
  frame16[(size_t)j * 256 + threadIdx.x] = (_Float16)v;
}

// ----------------------------- decoder GRU: one sequence of 16384, hidden = 80
__global__ void k_melgru(const float* __restrict__ gi, const float* __restrict__ Wh,
                         const float* __restrict__ bh, _Float16* __restrict__ melh) {
  __shared__ float h[80];
  __shared__ float gh[240];
  int tid = threadIdx.x;
  if (tid < 80) h[tid] = 0.f;
  __syncthreads();
  for (int s = 0; s < 16384; s++) {
    if (tid < 240) {
      float g = bh[tid];
      const float* w = Wh + (size_t)tid * 80;
      for (int c = 0; c < 80; c++) g += h[c] * w[c];
      gh[tid] = g;
    }
    __syncthreads();
    float hn = 0.f;
    bool act = tid < 80;
    if (act) {
      const float* gr = gi + (size_t)s * 240;
      float r = sigf(gr[tid] + gh[tid]);
      float z = sigf(gr[80 + tid] + gh[80 + tid]);
      float n = tanhf(gr[160 + tid] + r * gh[160 + tid]);
      hn = (1.f - z) * n + z * h[tid];
      melh[(size_t)s * 96 + tid] = (_Float16)hn;
    }
    __syncthreads();
    if (act) h[tid] = hn;
    __syncthreads();
  }
}

// ---------------------------------------------------------------------- launch
extern "C" void kernel_launch(void* const* d_in, const int* in_sizes, int n_in,
                              void* d_out, int out_size, void* d_ws, size_t ws_size,
                              hipStream_t stream) {
  const float* mel   = (const float*)d_in[0];
  const int*   len   = (const int*)  d_in[1];
  const float* encWi = (const float*)d_in[2];
  const float* encWh = (const float*)d_in[3];
  const float* encbi = (const float*)d_in[4];
  const float* encbh = (const float*)d_in[5];
  const float* projW = (const float*)d_in[6];
  const float* projb = (const float*)d_in[7];
  const float* convW = (const float*)d_in[8];
  const float* convb = (const float*)d_in[9];
  const float* melWi = (const float*)d_in[10];
  const float* melWh = (const float*)d_in[11];
  const float* melbi = (const float*)d_in[12];
  const float* melbh = (const float*)d_in[13];
  const float* outW  = (const float*)d_in[14];
  const float* outb  = (const float*)d_in[15];

  float* out = (float*)d_out;
  float* o_recon = out;                                     // [16,1024,80]
  float* o_frame = o_recon + (size_t)16 * 1024 * 80;        // [16384,256]
  float* o_seg   = o_frame + (size_t)16384 * 256;           // [16,1025,256]
  float* o_dur   = o_seg   + (size_t)16 * 1025 * 256;       // [16,1025]
  float* o_lat   = o_dur   + (size_t)16 * 1025;             // [16384,64]
  float* o_pb    = o_lat   + (size_t)16384 * 64;            // [16,1024]
  float* o_mask  = o_pb    + (size_t)16384;                 // [16,1024]
  float* o_d     = o_mask  + (size_t)16384;                 // [16,1024]
  float* o_preds = o_d     + (size_t)16384;                 // [16,1023,2]

  char* wp = (char*)d_ws;
  auto alloc = [&](size_t bytes) -> char* {
    char* p = wp; wp += (bytes + 255) & ~(size_t)255; return p;
  };
  _Float16* melf16  = (_Float16*)alloc((size_t)16384 * 96 * 2);
  _Float16* wi16    = (_Float16*)alloc((size_t)768 * 96 * 2);
  _Float16* wh16    = (_Float16*)alloc((size_t)768 * 256 * 2);
  _Float16* pW16    = (_Float16*)alloc((size_t)64 * 256 * 2);
  _Float16* mWi16   = (_Float16*)alloc((size_t)240 * 256 * 2);
  _Float16* oW16    = (_Float16*)alloc((size_t)80 * 96 * 2);
  _Float16* wc16    = (_Float16*)alloc((size_t)5 * 256 * 256 * 2);
  float*    gix     = (float*)   alloc((size_t)16384 * 768 * 4);   // reused as gi2
  float*    gru32   = (float*)   alloc((size_t)16384 * 256 * 4);
  _Float16* gru16   = (_Float16*)alloc((size_t)16384 * 256 * 2);
  _Float16* gt16    = (_Float16*)alloc((size_t)16 * 256 * 1024 * 2);
  float*    sim     = (float*)   alloc((size_t)16 * 1023 * 4);
  float*    smm     = (float*)   alloc((size_t)32 * 4);
  float*    dbuf    = (float*)   alloc((size_t)16384 * 4);
  float*    pbbuf   = (float*)   alloc((size_t)16384 * 4);
  float*    segacc  = (float*)   alloc((size_t)16384 * 4);
  int*      duri    = (int*)     alloc((size_t)16400 * 4);
  float*    wdiv    = (float*)   alloc((size_t)16400 * 4);
  int*      starts  = (int*)     alloc((size_t)16404 * 4);
  _Float16* segbuf  = (_Float16*)alloc((size_t)16 * 1048 * 256 * 2);
  _Float16* frame16 = (_Float16*)alloc((size_t)16384 * 256 * 2);
  _Float16* melh16  = (_Float16*)alloc((size_t)16384 * 96 * 2);
  float*    gi2     = gix;                                   // overlay (gix dead)

  // ---- weight / input precision staging
  k_cvt<<<6144, 256, 0, stream>>>(mel,   melf16, 16384, 80, 96);
  k_cvt<<<288,  256, 0, stream>>>(encWi, wi16,   768,  80, 96);
  k_cvt<<<768,  256, 0, stream>>>(encWh, wh16,   768, 256, 256);
  k_cvt<<<64,   256, 0, stream>>>(projW, pW16,   64,  256, 256);
  k_cvt<<<240,  256, 0, stream>>>(melWi, mWi16,  240, 256, 256);
  k_cvt<<<30,   256, 0, stream>>>(outW,  oW16,   80,  80, 96);
  k_convw<<<1280, 256, 0, stream>>>(convW, wc16);
  k_zero16<<<16768, 256, 0, stream>>>(segbuf, (size_t)16 * 1048 * 256);
  k_zero16<<<6144,  256, 0, stream>>>(melh16, (size_t)16384 * 96);

  // ---- encoder: gi = mel @ Wi^T + bi   [16384 x 768 x 96]
  k_gemm<4><<<dim3(12, 256), 128, 0, stream>>>(melf16, 96, wi16, 96, encbi,
                                               gix, 768, 16384, 768, 96);
  // ---- encoder recurrence (persistent, WMMA hidden GEMM)
  k_encgru<<<1, 512, 0, stream>>>(gix, wh16, encbh, gru32, gru16, gt16);

  // ---- latent = gru_out @ projW^T + b   [16384 x 64 x 256]
  k_gemm<4><<<dim3(1, 256), 128, 0, stream>>>(gru16, 256, pW16, 256, projb,
                                              o_lat, 64, 16384, 64, 256);

  // ---- CPC preds (latent) and boundary sim (gru_out)
  k_cos<<<2046, 256, 0, stream>>>(gru32, 256, 1024, 0, sim, nullptr);
  k_cos<<<2046, 256, 0, stream>>>(o_lat, 64, 1024, 1, nullptr, o_preds);

  // ---- boundaries
  k_minmax<<<16, 256, 0, stream>>>(sim, len, smm);
  k_d<<<64, 256, 0, stream>>>(sim, len, smm, dbuf, o_d, o_mask);
  k_pb<<<64, 256, 0, stream>>>(dbuf, len, pbbuf, o_pb);
  k_cumsum<<<1, 32, 0, stream>>>(pbbuf, len, segacc);
  k_dur<<<16400, 256, 0, stream>>>(segacc, duri, wdiv, o_dur);

  // ---- segment pooling (on-the-fly W, async-LDS staged B, WMMA), then conv1d
  k_seg<<<dim3(16, 9), 256, 0, stream>>>(segacc, wdiv, gt16, segbuf);
  k_conv<<<dim3(16, 9), 256, 0, stream>>>(segbuf, wc16, convb, o_seg);

  // ---- duration upsample
  k_prefix<<<1, 1, 0, stream>>>(duri, starts);
  k_up<<<16384, 256, 0, stream>>>(starts, o_seg, o_frame, frame16);

  // ---- decoder: gi2 = frame @ melWi^T + bi   [16384 x 240 x 256]
  k_gemm<5><<<dim3(3, 256), 128, 0, stream>>>(frame16, 256, mWi16, 256, melbi,
                                              gi2, 240, 16384, 240, 256);
  k_melgru<<<1, 256, 0, stream>>>(gi2, melWh, melbh, melh16);

  // ---- recon = mel_h @ outW^T + b   [16384 x 80 x 96]
  k_gemm<5><<<dim3(1, 256), 128, 0, stream>>>(melh16, 96, oW16, 96, outb,
                                              o_recon, 80, 16384, 80, 96);
}
